// SelfAttention_25890062860916
// MI455X (gfx1250) — compile-verified, matrix operand order checked by reading, not run on hardware
//
#include <hip/hip_runtime.h>

// ---------------------------------------------------------------------------
// Self-attention for MI455X (gfx1250): bf16 WMMA everywhere, flash-attention
// streaming so the (B,H,N,N) score tensor is never materialized.
// DPP16 softmax reductions (off the LDS pipe), async global->LDS tile fills
// (ASYNCcnt path) via __builtin_amdgcn_global_load_async_to_lds_b128.
// ---------------------------------------------------------------------------

typedef __attribute__((ext_vector_type(16))) __bf16 v16bf;
typedef __attribute__((ext_vector_type(8)))  __bf16 v8bf;
typedef __attribute__((ext_vector_type(8)))  float  v8f;

#define BB   4
#define HH_  8
#define NN   2048
#define DX   384
#define DQ   48
#define DQP  64                 // dq padded to 2x K=32 WMMA chunks
#define MTOT (BB * NN)          // 8192 rows when batch is folded into M
#define DV   (DX * HH_)         // 3072

// ---------------------------------------------------------------------------
// CDNA5 feature probes
// ---------------------------------------------------------------------------
#if __has_builtin(__builtin_amdgcn_global_load_async_to_lds_b128)
#define HAVE_ASYNC_LDS 1
#else
#define HAVE_ASYNC_LDS 0
#endif

#if HAVE_ASYNC_LDS
typedef int i32x4 __attribute__((vector_size(16)));
typedef __attribute__((address_space(1))) i32x4 as1_i32x4;
typedef __attribute__((address_space(3))) i32x4 as3_i32x4;
// 16B global -> LDS, no VGPR round trip; tracked by ASYNCcnt.
__device__ __forceinline__ void async_cp16(void* l, const void* g) {
  __builtin_amdgcn_global_load_async_to_lds_b128((as1_i32x4*)g, (as3_i32x4*)l, 0, 0);
}
__device__ __forceinline__ void wait_async0() {
#if __has_builtin(__builtin_amdgcn_s_wait_asynccnt)
  __builtin_amdgcn_s_wait_asynccnt(0);
#else
  asm volatile("s_wait_asynccnt 0x0" ::: "memory");
#endif
}
#endif

// DPP16 cross-lane reduction within a 16-lane half (all WMMA row stats live
// inside one half). Pure VALU: v_mov_b32_dpp + max/add, no ds_bpermute.
template <int CTRL>
__device__ __forceinline__ float dpp_mov(float x) {
  return __int_as_float(
      __builtin_amdgcn_update_dpp(0, __float_as_int(x), CTRL, 0xf, 0xf, true));
}
__device__ __forceinline__ float half16_max(float x) {
  x = fmaxf(x, dpp_mov<0xB1>(x));    // quad_perm(1,0,3,2)  : xor 1
  x = fmaxf(x, dpp_mov<0x4E>(x));    // quad_perm(2,3,0,1)  : xor 2
  x = fmaxf(x, dpp_mov<0x141>(x));   // row_half_mirror     : crosses 4-groups
  x = fmaxf(x, dpp_mov<0x140>(x));   // row_mirror          : crosses 8-groups
  return x;
}
__device__ __forceinline__ float half16_sum(float x) {
  x += dpp_mov<0xB1>(x);
  x += dpp_mov<0x4E>(x);
  x += dpp_mov<0x141>(x);
  x += dpp_mov<0x140>(x);
  return x;
}

// Build a v16bf A/B fragment from two 16-byte chunks (ISA 7.12.2 layout:
// lane = row%16, half = lane>>4 selects K subsets {8h..8h+7} and {16+8h..}).
__device__ __forceinline__ v16bf frag16(const __bf16* p0, const __bf16* p1) {
  v8bf lo = *(const v8bf*)p0;
  v8bf hi = *(const v8bf*)p1;
  return __builtin_shufflevector(lo, hi, 0,1,2,3,4,5,6,7,8,9,10,11,12,13,14,15);
}

__device__ __forceinline__ v8f wmma_bf16(v16bf a, v16bf b, v8f c) {
  // (neg_a, A, neg_b, B, c_mod, C, reuse_a, reuse_b)
  return __builtin_amdgcn_wmma_f32_16x16x32_bf16(false, a, false, b,
                                                 (short)0, c, false, false);
}

// ---------------------------------------------------------------------------
// Elementwise helpers
// ---------------------------------------------------------------------------
__global__ __launch_bounds__(256) void cvt_f32_bf16(const float* __restrict__ x,
                                                    __bf16* __restrict__ y, int n) {
  int i = blockIdx.x * 256 + threadIdx.x;
  if (i < n) y[i] = (__bf16)x[i];
}

__global__ __launch_bounds__(256) void zero_u32(unsigned int* __restrict__ p, int n) {
  int i = blockIdx.x * 256 + threadIdx.x;
  if (i < n) p[i] = 0u;
}

// ---------------------------------------------------------------------------
// Tiled GEMM: C[m,n] = sum_k A[m,k] * B[n,k]   (A: MxK row-major, B: NxK row-major)
// Block tile 128x128, BK=32, 8 waves in 2(M)x4(N), each wave 64x32 = 4x2 WMMA accums.
// MODE selects the scatter epilogue:
//   0: Q -> bf16, head-padded [(b*H+h)*N + row]*64 + d, scaled by 1/sqrt(48)
//   1: K -> bf16, same layout, unscaled
//   2: V -> bf16, feature-major [(b*H+h)*384 + d]*N + row
//   3: out -> fp32 row-major [m*384 + n]
// ---------------------------------------------------------------------------
template <int MODE>
__global__ __launch_bounds__(256) void gemm_bt(const __bf16* __restrict__ A,
                                               const __bf16* __restrict__ Bm,
                                               void* __restrict__ C,
                                               int K, int lda, int ldb) {
  __shared__ __align__(16) __bf16 As[128 * 40];   // stride 40 elems = 80B (16B aligned rows)
  __shared__ __align__(16) __bf16 Bs[128 * 40];

  const int tid  = threadIdx.x;
  const int wave = tid >> 5;
  const int lane = tid & 31;
  const int lr   = lane & 15;     // row within 16
  const int hf   = lane >> 4;     // K-half selector
  const int m0   = blockIdx.x * 128;
  const int n0   = blockIdx.y * 128;
  const int wm   = (wave & 1) * 64;
  const int wn   = (wave >> 1) * 32;

  const v8f zv = {0.f, 0.f, 0.f, 0.f, 0.f, 0.f, 0.f, 0.f};
  v8f acc[4][2];
#pragma unroll
  for (int i = 0; i < 4; ++i)
#pragma unroll
    for (int j = 0; j < 2; ++j) acc[i][j] = zv;

  for (int k0 = 0; k0 < K; k0 += 32) {
    __syncthreads();
#pragma unroll
    for (int t = 0; t < 2; ++t) {                 // 512 16B chunks over 256 threads
      int c = tid + t * 256;
      int r = c >> 2, cc = (c & 3) << 3;
#if HAVE_ASYNC_LDS
      async_cp16(&As[r * 40 + cc], &A[(size_t)(m0 + r) * lda + k0 + cc]);
      async_cp16(&Bs[r * 40 + cc], &Bm[(size_t)(n0 + r) * ldb + k0 + cc]);
#else
      *(v8bf*)&As[r * 40 + cc] = *(const v8bf*)&A[(size_t)(m0 + r) * lda + k0 + cc];
      *(v8bf*)&Bs[r * 40 + cc] = *(const v8bf*)&Bm[(size_t)(n0 + r) * ldb + k0 + cc];
#endif
    }
#if HAVE_ASYNC_LDS
    wait_async0();
#endif
    __syncthreads();

    if (k0 + 32 < K) {                            // uniform branch; prefetch next K-slab
      __builtin_prefetch(&A[(size_t)(m0 + ((tid >> 1) & 127)) * lda + k0 + 32], 0, 1);
      __builtin_prefetch(&Bm[(size_t)(n0 + ((tid >> 1) & 127)) * ldb + k0 + 32], 0, 1);
    }

    v16bf af[4], bfm[2];
#pragma unroll
    for (int i = 0; i < 4; ++i)
      af[i] = frag16(&As[(wm + i * 16 + lr) * 40 + hf * 8],
                     &As[(wm + i * 16 + lr) * 40 + 16 + hf * 8]);
#pragma unroll
    for (int j = 0; j < 2; ++j)
      bfm[j] = frag16(&Bs[(wn + j * 16 + lr) * 40 + hf * 8],
                      &Bs[(wn + j * 16 + lr) * 40 + 16 + hf * 8]);
#pragma unroll
    for (int i = 0; i < 4; ++i)
#pragma unroll
      for (int j = 0; j < 2; ++j)
        acc[i][j] = wmma_bf16(af[i], bfm[j], acc[i][j]);
  }

  // Epilogue: C/D layout -> vgpr v holds row (v + 8*hf), col = lane&15.
#pragma unroll
  for (int i = 0; i < 4; ++i)
#pragma unroll
    for (int j = 0; j < 2; ++j)
#pragma unroll
      for (int v = 0; v < 8; ++v) {
        int m = m0 + wm + i * 16 + v + 8 * hf;
        int n = n0 + wn + j * 16 + lr;
        float val = acc[i][j][v];
        if constexpr (MODE == 0 || MODE == 1) {
          if constexpr (MODE == 0) val *= 0.14433756729740643f;  // 1/sqrt(48)
          int bb = m >> 11, row = m & (NN - 1);
          int hh = n / DQ, d = n - hh * DQ;
          ((__bf16*)C)[(((size_t)(bb * HH_ + hh) * NN + row) << 6) + d] = (__bf16)val;
        } else if constexpr (MODE == 2) {
          int bb = m >> 11, row = m & (NN - 1);
          int hh = n / DX, d = n - hh * DX;
          ((__bf16*)C)[((size_t)((bb * HH_ + hh) * DX + d)) * NN + row] = (__bf16)val;
        } else {
          ((float*)C)[(size_t)m * DX + n] = val;
        }
      }
}

// ---------------------------------------------------------------------------
// Flash attention: grid (N/64, H, B), 256 threads = 8 waves.
// Wave w: query group g = w&3 (16 queries), feature half fh = w>>2 (192 of 384).
// Streams 64-key blocks: S = Q K^T (4 tiles), online softmax via DPP16
// reductions inside the 16-lane WMMA halves, P re-striped via per-wave LDS,
// R += P*V. Writes bf16 R into Rt[(b*N+row)*3072 + h*384 + f].
// ---------------------------------------------------------------------------
__global__ __launch_bounds__(256) void flash_attn(const __bf16* __restrict__ Qp,
                                                  const __bf16* __restrict__ Kp,
                                                  const __bf16* __restrict__ Vt,
                                                  __bf16* __restrict__ Rt) {
  __shared__ __align__(16) __bf16 Ks[64 * 72];        // 64 keys x 64 dq (stride 72)
  __shared__ __align__(16) __bf16 Vs[384 * 72];       // 384 features x 64 keys
  __shared__ __align__(16) __bf16 Ps[8][16 * 72];     // per-wave P tile (16 x 64)

  const int tid  = threadIdx.x;
  const int wave = tid >> 5;
  const int lane = tid & 31;
  const int lr   = lane & 15;
  const int hf   = lane >> 4;
  const int h    = blockIdx.y;
  const int b    = blockIdx.z;
  const int bh   = b * HH_ + h;
  const int g    = wave & 3;
  const int fh   = wave >> 2;
  const int q0   = blockIdx.x * 64 + g * 16;          // absolute query base for wave

  // Preload Q fragments (dq=64 -> two K=32 chunks), directly from global.
  const __bf16* Qbase = Qp + ((size_t)bh * NN + q0) * DQP;
  v16bf qf[2];
#pragma unroll
  for (int c = 0; c < 2; ++c)
    qf[c] = frag16(&Qbase[lr * DQP + c * 32 + hf * 8],
                   &Qbase[lr * DQP + c * 32 + 16 + hf * 8]);

  const v8f zv = {0.f, 0.f, 0.f, 0.f, 0.f, 0.f, 0.f, 0.f};
  v8f racc[12];
#pragma unroll
  for (int t = 0; t < 12; ++t) racc[t] = zv;
  float mrow[8], lrow[8];
#pragma unroll
  for (int v = 0; v < 8; ++v) { mrow[v] = -1e30f; lrow[v] = 0.f; }

  for (int kb = 0; kb < NN / 64; ++kb) {
    const __bf16* Kbase = Kp + ((size_t)bh * NN + kb * 64) * DQP;
    const __bf16* Vbase = Vt + (size_t)bh * DX * NN + (size_t)kb * 64;

    __syncthreads();
#pragma unroll
    for (int t = 0; t < 2; ++t) {                     // K block: 64x64 bf16
      int c = tid + t * 256;
      int r = c >> 3, cc = (c & 7) << 3;
#if HAVE_ASYNC_LDS
      async_cp16(&Ks[r * 72 + cc], &Kbase[r * DQP + cc]);
#else
      *(v8bf*)&Ks[r * 72 + cc] = *(const v8bf*)&Kbase[r * DQP + cc];
#endif
    }
#pragma unroll
    for (int t = 0; t < 12; ++t) {                    // V block: 384x64 bf16
      int c = tid + t * 256;
      int r = c >> 3, cc = (c & 7) << 3;
#if HAVE_ASYNC_LDS
      async_cp16(&Vs[r * 72 + cc], &Vbase[(size_t)r * NN + cc]);
#else
      *(v8bf*)&Vs[r * 72 + cc] = *(const v8bf*)&Vbase[(size_t)r * NN + cc];
#endif
    }
#if HAVE_ASYNC_LDS
    wait_async0();
#endif
    __syncthreads();

    if (kb + 1 < NN / 64) {                           // uniform; prefetch next block
      __builtin_prefetch(Kbase + 64 * DQP + (tid << 4), 0, 1);
      __builtin_prefetch(Vbase + 64 + (size_t)tid * NN, 0, 1);
    }

    // S = Q K^T : 16 queries x 64 keys = 4 WMMA tiles x 2 K-chunks
    v8f s[4];
#pragma unroll
    for (int j = 0; j < 4; ++j) s[j] = zv;
#pragma unroll
    for (int j = 0; j < 4; ++j)
#pragma unroll
      for (int c = 0; c < 2; ++c) {
        v16bf kf = frag16(&Ks[(j * 16 + lr) * 72 + c * 32 + hf * 8],
                          &Ks[(j * 16 + lr) * 72 + c * 32 + 16 + hf * 8]);
        s[j] = wmma_bf16(qf[c], kf, s[j]);
      }

    // Online softmax. Row (v + 8*hf) lives across the lanes of one 16-lane
    // half; DPP16 reductions keep this entirely in the VALU pipe.
    float scale[8];
#pragma unroll
    for (int v = 0; v < 8; ++v) {
      float mx = fmaxf(fmaxf(s[0][v], s[1][v]), fmaxf(s[2][v], s[3][v]));
      mx = half16_max(mx);
      float mn = fmaxf(mrow[v], mx);
      scale[v] = __expf(mrow[v] - mn);
      float sum = 0.f;
#pragma unroll
      for (int j = 0; j < 4; ++j) {
        float e = __expf(s[j][v] - mn);
        s[j][v] = e;
        sum += e;
      }
      sum = half16_sum(sum);
      lrow[v] = lrow[v] * scale[v] + sum;
      mrow[v] = mn;
    }
#pragma unroll
    for (int t = 0; t < 12; ++t)
#pragma unroll
      for (int v = 0; v < 8; ++v) racc[t][v] *= scale[v];

    // Re-stripe P from C/D layout to A layout through per-wave LDS.
    __bf16* myP = &Ps[wave][0];
#pragma unroll
    for (int j = 0; j < 4; ++j)
#pragma unroll
      for (int v = 0; v < 8; ++v)
        myP[(v + 8 * hf) * 72 + j * 16 + lr] = (__bf16)s[j][v];

    // R += P * V over this wave's 192 features.
#pragma unroll
    for (int c = 0; c < 2; ++c) {
      v16bf pf = frag16(&myP[lr * 72 + c * 32 + hf * 8],
                        &myP[lr * 72 + c * 32 + 16 + hf * 8]);
#pragma unroll
      for (int t = 0; t < 12; ++t) {
        int f = fh * 192 + t * 16;
        v16bf vf = frag16(&Vs[(f + lr) * 72 + c * 32 + hf * 8],
                          &Vs[(f + lr) * 72 + c * 32 + 16 + hf * 8]);
        racc[t] = wmma_bf16(pf, vf, racc[t]);
      }
    }
  }

  // Normalize (8 reciprocals, not 96 divides) and store bf16 R rows.
  float inv[8];
#pragma unroll
  for (int v = 0; v < 8; ++v) inv[v] = 1.0f / lrow[v];
#pragma unroll
  for (int t = 0; t < 12; ++t) {
    int col = h * DX + fh * 192 + t * 16 + lr;
#pragma unroll
    for (int v = 0; v < 8; ++v) {
      int row = q0 + v + 8 * hf;
      Rt[((size_t)(b * NN + row)) * DV + col] = (__bf16)(racc[t][v] * inv[v]);
    }
  }
}

// ---------------------------------------------------------------------------
// Host-side launcher
// ---------------------------------------------------------------------------
extern "C" void kernel_launch(void* const* d_in, const int* in_sizes, int n_in,
                              void* d_out, int out_size, void* d_ws, size_t ws_size,
                              hipStream_t stream) {
  const float* X  = (const float*)d_in[0];
  const float* Wq = (const float*)d_in[1];
  const float* Wk = (const float*)d_in[2];
  const float* Wv = (const float*)d_in[3];
  const float* Wr = (const float*)d_in[4];

  char* p = (char*)d_ws;
  __bf16* Xb  = (__bf16*)p; p += (size_t)MTOT * DX * 2;
  __bf16* Wqb = (__bf16*)p; p += (size_t)DX * DX * 2;
  __bf16* Wkb = (__bf16*)p; p += (size_t)DX * DX * 2;
  __bf16* Wvb = (__bf16*)p; p += (size_t)DV * DX * 2;
  __bf16* Wrb = (__bf16*)p; p += (size_t)DX * DV * 2;
  __bf16* Qp  = (__bf16*)p; p += (size_t)BB * HH_ * NN * DQP * 2;
  __bf16* Kp  = (__bf16*)p; p += (size_t)BB * HH_ * NN * DQP * 2;
  __bf16* Vt  = (__bf16*)p; p += (size_t)BB * HH_ * DX * NN * 2;
  __bf16* Rt  = (__bf16*)p;

  // fp32 -> bf16 conversions
  {
    int n;
    n = MTOT * DX; cvt_f32_bf16<<<dim3((n + 255) / 256), 256, 0, stream>>>(X,  Xb,  n);
    n = DX * DX;   cvt_f32_bf16<<<dim3((n + 255) / 256), 256, 0, stream>>>(Wq, Wqb, n);
    n = DX * DX;   cvt_f32_bf16<<<dim3((n + 255) / 256), 256, 0, stream>>>(Wk, Wkb, n);
    n = DV * DX;   cvt_f32_bf16<<<dim3((n + 255) / 256), 256, 0, stream>>>(Wv, Wvb, n);
    n = DX * DV;   cvt_f32_bf16<<<dim3((n + 255) / 256), 256, 0, stream>>>(Wr, Wrb, n);
  }
  // Zero head-padded Q/K (pad lanes 48..63 must be 0 for the padded QK^T).
  {
    int zw = BB * HH_ * NN * DQP / 2;  // bf16 pairs as u32 words
    zero_u32<<<dim3((zw + 255) / 256), 256, 0, stream>>>((unsigned int*)Qp, zw);
    zero_u32<<<dim3((zw + 255) / 256), 256, 0, stream>>>((unsigned int*)Kp, zw);
  }

  // Projections (batch folded into M = 8192)
  gemm_bt<0><<<dim3(MTOT / 128, DX / 128), 256, 0, stream>>>(Xb, Wqb, Qp, DX, DX, DX);
  gemm_bt<1><<<dim3(MTOT / 128, DX / 128), 256, 0, stream>>>(Xb, Wkb, Kp, DX, DX, DX);
  gemm_bt<2><<<dim3(MTOT / 128, DV / 128), 256, 0, stream>>>(Xb, Wvb, Vt, DX, DX, DX);

  // Streaming softmax attention
  flash_attn<<<dim3(NN / 64, HH_, BB), 256, 0, stream>>>(Qp, Kp, Vt, Rt);

  // Output projection -> fp32 d_out (B, N, 384)
  gemm_bt<3><<<dim3(MTOT / 128, DX / 128), 256, 0, stream>>>(Rt, Wrb, d_out, DV, DV, DV);
}
